// TripleLoss_80736795230438
// MI455X (gfx1250) — matrix operand order repaired and checked by reading, not compiled, not run on hardware
//
#include <hip/hip_runtime.h>

typedef _Float16 v16h __attribute__((ext_vector_type(16)));
typedef _Float16 v8h  __attribute__((ext_vector_type(8)));
typedef _Float16 v4h  __attribute__((ext_vector_type(4)));
typedef float    v8f  __attribute__((ext_vector_type(8)));

#define NROWS   8192
#define DIM     128
#define CHUNKS  16
#define TILES_PER_CHUNK 32   // 16 chunks * 32 tiles * 16 cols = 8192 columns
#define COLS_PER_CHUNK  (TILES_PER_CHUNK * 16)   // 512
#define WAVES_PER_BLOCK 8
#define BPAD    8            // LDS row pad (halves): stride 136 -> conflict-free
#define BSTRIDE (DIM + BPAD)
#define MARGIN  0.3f

// ---------------------------------------------------------------------------
// Kernel 1: L2-normalize each row and convert to f16 (row-major [N, D]).
// One wave per row; each lane owns 4 contiguous floats.
// ---------------------------------------------------------------------------
__global__ void tl_normalize(const float* __restrict__ x,
                             _Float16* __restrict__ xh) {
  const int wave = threadIdx.x >> 5;
  const int lane = threadIdx.x & 31;
  const int row  = blockIdx.x * 8 + wave;

  const float4 v = *(const float4*)(x + (size_t)row * DIM + lane * 4);
  float s = v.x * v.x + v.y * v.y + v.z * v.z + v.w * v.w;
  s += __shfl_xor(s, 1);
  s += __shfl_xor(s, 2);
  s += __shfl_xor(s, 4);
  s += __shfl_xor(s, 8);
  s += __shfl_xor(s, 16);
  const float inv = 1.0f / (sqrtf(s) + 1e-12f);

  v4h h;
  h.x = (_Float16)(v.x * inv);
  h.y = (_Float16)(v.y * inv);
  h.z = (_Float16)(v.z * inv);
  h.w = (_Float16)(v.w * inv);
  *(v4h*)(xh + (size_t)row * DIM + lane * 4) = h;
}

// ---------------------------------------------------------------------------
// Kernel 2: fused GEMM-tile + hard mining, B staged through LDS.
// Block = 8 waves sharing one 512-column chunk; wave w owns row tile
// (rowGroup*8 + w). Column tiles are copied global->LDS once per block
// (double-buffered) and every wave builds its B fragments from LDS.
// For each col tile: 4x v_wmma_f32_16x16x32_f16 (K=128), then fold the C tile
// into per-row running min (hardest positive) / max (hardest negative).
// ---------------------------------------------------------------------------
__global__ void tl_mine(const _Float16* __restrict__ xh,
                        const int* __restrict__ lab,
                        float* __restrict__ pap,
                        float* __restrict__ pan) {
  __shared__ _Float16 sB[2][16 * BSTRIDE];   // 2 x 16 x 136 halves = 8.5 KB
  __shared__ int      sLab[COLS_PER_CHUNK];  // 2 KB

  const int tid     = threadIdx.x;
  const int lane    = tid & 31;
  const int waveId  = tid >> 5;                    // 0..7
  const int chunk   = blockIdx.x & (CHUNKS - 1);   // 0..15
  const int rowGrp  = blockIdx.x >> 4;             // 0..63
  const int rowTile = rowGrp * WAVES_PER_BLOCK + waveId;
  const int rowbase = rowTile * 16;
  const int laneN   = lane & 15;
  const int half    = lane >> 4;  // 0: rows v, K 0-7,16-23 | 1: rows v+8, K 8-15,24-31
  const int colBase = chunk * COLS_PER_CHUNK;

  // Tile-copy role: thread tid moves 16 bytes (8 halves) of row `cRow`.
  const int cRow = tid >> 4;      // 0..15 (tile column index)
  const int cSeg = tid & 15;      // 0..15 (8-half segment within the row)

  // Stage this chunk's labels once.
  sLab[tid]       = lab[colBase + tid];
  sLab[tid + 256] = lab[colBase + tid + 256];

  // A fragments: row M = rowbase + laneN; per K-tile two contiguous 8-half runs.
  const _Float16* rowPtr = xh + (size_t)(rowbase + laneN) * DIM;
  v16h a[4];
#pragma unroll
  for (int kt = 0; kt < 4; ++kt) {
    v8h lo = *(const v8h*)(rowPtr + kt * 32 + half * 8);
    v8h hi = *(const v8h*)(rowPtr + kt * 32 + 16 + half * 8);
    a[kt] = __builtin_shufflevector(lo, hi,
                                    0, 1, 2, 3, 4, 5, 6, 7,
                                    8, 9, 10, 11, 12, 13, 14, 15);
  }

  int labR[8];
#pragma unroll
  for (int v = 0; v < 8; ++v) labR[v] = lab[rowbase + half * 8 + v];

  float ap[8], an[8];
#pragma unroll
  for (int v = 0; v < 8; ++v) { ap[v] = 3.0e38f; an[v] = -3.0e38f; }

  // Prologue: stage tile 0.
  *(v8h*)(&sB[0][cRow * BSTRIDE + cSeg * 8]) =
      *(const v8h*)(xh + (size_t)(colBase + cRow) * DIM + cSeg * 8);
  __syncthreads();

  int cur = 0;
  for (int t = 0; t < TILES_PER_CHUNK; ++t) {
    const int nxt = cur ^ 1;

    // Issue next tile's global loads early (latency hidden behind WMMA).
    v8h pre;
    const bool havePre = (t + 1 < TILES_PER_CHUNK);
    if (havePre)
      pre = *(const v8h*)(xh + (size_t)(colBase + (t + 1) * 16 + cRow) * DIM +
                          cSeg * 8);

    // B fragment: column N = laneN of this tile; per K-tile two 8-half runs
    // (half*8, 16+half*8) at sB row laneN (stride 136 halves -> conflict-free banks).
    const _Float16* bRow = &sB[cur][laneN * BSTRIDE];
    v8f c = {};
#pragma unroll
    for (int kt = 0; kt < 4; ++kt) {
      v8h blo = *(const v8h*)(bRow + kt * 32 + half * 8);
      v8h bhi = *(const v8h*)(bRow + kt * 32 + 16 + half * 8);
      v16h b = __builtin_shufflevector(blo, bhi,
                                       0, 1, 2, 3, 4, 5, 6, 7,
                                       8, 9, 10, 11, 12, 13, 14, 15);
      c = __builtin_amdgcn_wmma_f32_16x16x32_f16(
              /*neg_a=*/false, a[kt], /*neg_b=*/false, b,
              /*c_mod=*/(short)0, c, /*reuse_a=*/false, /*reuse_b=*/false);
    }

    const int gcol = colBase + t * 16 + laneN;
    const int labC = sLab[t * 16 + laneN];
#pragma unroll
    for (int v = 0; v < 8; ++v) {
      const float cv    = c[v];
      const bool  same  = (labR[v] == labC);
      const int   grow  = rowbase + half * 8 + v;
      const bool  ispos = same && (grow != gcol);
      const float apc   = ispos ? cv : (cv + 2.0f);  // where(is_pos, d, d+2)
      const float anc   = same ? (cv - 2.0f) : cv;   // where(same, d-2, d)
      ap[v] = fminf(ap[v], apc);
      an[v] = fmaxf(an[v], anc);
    }

    if (havePre)
      *(v8h*)(&sB[nxt][cRow * BSTRIDE + cSeg * 8]) = pre;
    __syncthreads();
    cur = nxt;
  }

  // Reduce over the 16 lanes striping the N dimension (xor masks < 16 stay
  // inside each half-wave, so rows v and v+8 reduce independently).
#pragma unroll
  for (int v = 0; v < 8; ++v) {
#pragma unroll
    for (int m = 1; m < 16; m <<= 1) {
      ap[v] = fminf(ap[v], __shfl_xor(ap[v], m));
      an[v] = fmaxf(an[v], __shfl_xor(an[v], m));
    }
  }

  if (laneN == 0) {
#pragma unroll
    for (int v = 0; v < 8; ++v) {
      const int grow = rowbase + half * 8 + v;
      pap[grow * CHUNKS + chunk] = ap[v];
      pan[grow * CHUNKS + chunk] = an[v];
    }
  }
}

// ---------------------------------------------------------------------------
// Kernel 3: per-row combine over chunks, then deterministic single-block sum.
// ---------------------------------------------------------------------------
__global__ void tl_finalize(const float* __restrict__ pap,
                            const float* __restrict__ pan,
                            float* __restrict__ out) {
  __shared__ float red[1024];
  const int tid = threadIdx.x;
  float sum = 0.0f;
  for (int r = tid; r < NROWS; r += 1024) {
    float ap = 3.0e38f, an = -3.0e38f;
#pragma unroll
    for (int c = 0; c < CHUNKS; ++c) {
      ap = fminf(ap, pap[r * CHUNKS + c]);
      an = fmaxf(an, pan[r * CHUNKS + c]);
    }
    sum += an + MARGIN - ap;
  }
  red[tid] = sum;
  __syncthreads();
  for (int s = 512; s > 0; s >>= 1) {
    if (tid < s) red[tid] += red[tid + s];
    __syncthreads();
  }
  if (tid == 0) out[0] = red[0];
}

// ---------------------------------------------------------------------------
extern "C" void kernel_launch(void* const* d_in, const int* in_sizes, int n_in,
                              void* d_out, int out_size, void* d_ws, size_t ws_size,
                              hipStream_t stream) {
  (void)in_sizes; (void)n_in; (void)out_size; (void)ws_size;
  const float* x      = (const float*)d_in[0];
  const int*   labels = (const int*)d_in[1];
  float*       out    = (float*)d_out;

  char* ws = (char*)d_ws;
  _Float16* xh  = (_Float16*)ws;                                   // 2 MB
  float*    pap = (float*)(ws + (size_t)NROWS * DIM * sizeof(_Float16));
  float*    pan = pap + (size_t)NROWS * CHUNKS;                    // 512 KB each

  // 1) normalize + f16 convert: 8 waves/block, one wave per row.
  tl_normalize<<<NROWS / 8, 256, 0, stream>>>(x, xh);

  // 2) fused WMMA + mining: 64 row-groups (8 row tiles each) x 16 chunks,
  //    256 threads = 8 waves per block sharing B tiles via LDS.
  tl_mine<<<64 * CHUNKS, 256, 0, stream>>>(xh, labels, pap, pan);

  // 3) final reduction (single block, deterministic).
  tl_finalize<<<1, 1024, 0, stream>>>(pap, pan, out);
}
